// PairformerLayer_75900662055651
// MI455X (gfx1250) — matrix operand order repaired and checked by reading, not compiled, not run on hardware
//
#include <hip/hip_runtime.h>
#include <math.h>

typedef __attribute__((ext_vector_type(16))) _Float16 v16h;
typedef __attribute__((ext_vector_type(8)))  float    v8f;

#define NRES 256
#define PD   65536   /* 256*256 pair rows */
#define CZ   128
#define CS   384

__device__ __forceinline__ float fast_sigmoid(float x) {
  return __builtin_amdgcn_rcpf(1.0f + __expf(-x));   // v_rcp_f32, no div chain
}

// ---------------------------------------------------------------------------
// Strided batched GEMM on v_wmma_f32_16x16x32_f16.
//   C = alpha * (A x B) + bias + beta * C     (per batch)
// batch bz decomposed b1 = bz/S2, b2 = bz%S2, two batch strides per operand.
// All offsets fit in int32. No inner-loop guards anywhere: caller guarantees
// K%32==0 and memory-safe over-reads; garbage rows/cols are store-masked.
// A is always K-contiguous -> 4x float4 loads per fragment.
// ---------------------------------------------------------------------------
struct GP {
  const float* A; const float* B; float* C; const float* bias;
  int M, N, K, S2;
  int sAm, sAk, sBk, sBn, sCm, sCn, sbm, sbn;
  int bA1, bA2, bB1, bB2, bC1, bC2, bb1, bb2;
  float alpha, beta;
};

__device__ __forceinline__ int b_kidx(int h, int g) {
  // B k order used here (contiguous, same as A): lanes 0-15 hold K=0..15, lanes 16-31 hold K=16..31
  return g * 16 + h;
}
__device__ __forceinline__ void put4(v16h& v, int i, float4 f) {
  v[i]     = (_Float16)f.x;
  v[i + 1] = (_Float16)f.y;
  v[i + 2] = (_Float16)f.z;
  v[i + 3] = (_Float16)f.w;
}
// A fragment (16-bit A 16x32): lanes 0-15: K0..15, lanes 16-31: K16..31 (matches B)
__device__ __forceinline__ void load_afrag(v16h& af, const float* Arow, int k0, int g) {
  put4(af, 0,  *(const float4*)(Arow + k0 + g * 16));
  put4(af, 4,  *(const float4*)(Arow + k0 + g * 16 + 4));
  put4(af, 8,  *(const float4*)(Arow + k0 + g * 16 + 8));
  put4(af, 12, *(const float4*)(Arow + k0 + g * 16 + 12));
}
__device__ __forceinline__ void load_bfrag_k1(v16h& bf, const float* Bcol) {
  put4(bf, 0,  *(const float4*)(Bcol));
  put4(bf, 4,  *(const float4*)(Bcol + 4));
  put4(bf, 8,  *(const float4*)(Bcol + 8));
  put4(bf, 12, *(const float4*)(Bcol + 12));
}

// -------------------- 1x1 tile (small-N launches) --------------------
template <bool BK1>
__global__ __launch_bounds__(128) void gemm_fast(GP p) {
  const int wid  = threadIdx.x >> 5;
  const int lane = threadIdx.x & 31;
  const int mt   = blockIdx.y * 4 + wid;
  const int nt   = blockIdx.x;
  if (mt * 16 >= p.M) return;
  const int b1 = blockIdx.z / p.S2;
  const int b2 = blockIdx.z % p.S2;
  const int g  = lane >> 4;
  const int lm = lane & 15;
  const int m  = mt * 16 + lm;
  const int n  = nt * 16 + lm;
  const float* Arow = p.A + b1 * p.bA1 + b2 * p.bA2 + m * p.sAm;  // sAk == 1
  const float* Bb   = p.B + b1 * p.bB1 + b2 * p.bB2;
  float*       C    = p.C + b1 * p.bC1 + b2 * p.bC2;
  v8f acc = {};
  for (int k0 = 0; k0 < p.K; k0 += 32) {
    v16h af, bf;
    load_afrag(af, Arow, k0, g);
    if constexpr (BK1) {
      load_bfrag_k1(bf, Bb + n * p.sBn + k0 + g * 16);
    } else {
#pragma unroll
      for (int h = 0; h < 16; ++h)
        bf[h] = (_Float16)Bb[(k0 + b_kidx(h, g)) * p.sBk + n * p.sBn];
    }
    acc = __builtin_amdgcn_wmma_f32_16x16x32_f16(false, af, false, bf, (short)0, acc,
                                                 false, false);
  }
  const float* bb = p.bias ? (p.bias + b1 * p.bb1 + b2 * p.bb2) : nullptr;
#pragma unroll
  for (int r = 0; r < 8; ++r) {
    const int mo = mt * 16 + r + 8 * g;   // C/D: VGPR r -> row r (+8 upper half-wave)
    const int no = nt * 16 + lm;
    if (mo < p.M && no < p.N) {
      const int ci = mo * p.sCm + no * p.sCn;
      float v = p.alpha * acc[r];
      if (bb) v += bb[mo * p.sbm + no * p.sbn];
      if (p.beta != 0.0f) v += p.beta * C[ci];
      C[ci] = v;
    }
  }
}

// -------------------- 2x2 tile (bulk launches, M%32==0, N%32==0-safe) ------
template <bool BK1>
__global__ __launch_bounds__(128) void gemm_fast2(GP p) {
  const int wid  = threadIdx.x >> 5;
  const int lane = threadIdx.x & 31;
  const int mt   = (blockIdx.y * 4 + wid) * 2;   // first of a 16-row tile pair
  const int nt   = blockIdx.x * 2;
  if (mt * 16 >= p.M) return;
  const int b1 = blockIdx.z / p.S2;
  const int b2 = blockIdx.z % p.S2;
  const int g  = lane >> 4;
  const int lm = lane & 15;
  const int m0 = mt * 16 + lm;
  const int n0 = nt * 16 + lm;
  const float* Arow0 = p.A + b1 * p.bA1 + b2 * p.bA2 + m0 * p.sAm;
  const float* Arow1 = Arow0 + 16 * p.sAm;
  const float* Bb    = p.B + b1 * p.bB1 + b2 * p.bB2;
  float*       C     = p.C + b1 * p.bC1 + b2 * p.bC2;
  v8f acc00 = {}, acc01 = {}, acc10 = {}, acc11 = {};
  for (int k0 = 0; k0 < p.K; k0 += 32) {
    // prefetch next k-tile of A (speculative; harmless past the end)
    __builtin_prefetch(Arow0 + k0 + 32, 0, 3);
    __builtin_prefetch(Arow1 + k0 + 32, 0, 3);
    v16h a0, a1, bf0, bf1;
    load_afrag(a0, Arow0, k0, g);
    load_afrag(a1, Arow1, k0, g);
    if constexpr (BK1) {
      const float* Bcol0 = Bb + n0 * p.sBn + k0 + g * 16;
      load_bfrag_k1(bf0, Bcol0);
      load_bfrag_k1(bf1, Bcol0 + 16 * p.sBn);
    } else {
#pragma unroll
      for (int h = 0; h < 16; ++h) {
        const int ko = (k0 + b_kidx(h, g)) * p.sBk + n0 * p.sBn;
        bf0[h] = (_Float16)Bb[ko];
        bf1[h] = (_Float16)Bb[ko + 16 * p.sBn];
      }
    }
    acc00 = __builtin_amdgcn_wmma_f32_16x16x32_f16(false, a0, false, bf0, (short)0, acc00, false, false);
    acc01 = __builtin_amdgcn_wmma_f32_16x16x32_f16(false, a0, false, bf1, (short)0, acc01, false, false);
    acc10 = __builtin_amdgcn_wmma_f32_16x16x32_f16(false, a1, false, bf0, (short)0, acc10, false, false);
    acc11 = __builtin_amdgcn_wmma_f32_16x16x32_f16(false, a1, false, bf1, (short)0, acc11, false, false);
  }
  const float* bb = p.bias ? (p.bias + b1 * p.bb1 + b2 * p.bb2) : nullptr;
  auto store_tile = [&](const v8f& acc, int mtt, int ntt) {
#pragma unroll
    for (int r = 0; r < 8; ++r) {
      const int mo = mtt * 16 + r + 8 * g;
      const int no = ntt * 16 + lm;
      if (mo < p.M && no < p.N) {
        const int ci = mo * p.sCm + no * p.sCn;
        float v = p.alpha * acc[r];
        if (bb) v += bb[mo * p.sbm + no * p.sbn];
        if (p.beta != 0.0f) v += p.beta * C[ci];
        C[ci] = v;
      }
    }
  };
  store_tile(acc00, mt, nt);
  store_tile(acc01, mt, nt + 1);
  store_tile(acc10, mt + 1, nt);
  store_tile(acc11, mt + 1, nt + 1);
}

// ---------------------------------------------------------------------------
// Fused SwiGLU dual GEMM: C = silu(A @ W1) * (A @ W2).  A fragment loaded once
// feeds two WMMA chains. Row-major A[MxK], W[KxN]; M%16==0, N%16==0, K%32==0.
// ---------------------------------------------------------------------------
__global__ __launch_bounds__(128) void dualgemm_silu(const float* __restrict__ A,
                                                     const float* __restrict__ W1,
                                                     const float* __restrict__ W2,
                                                     float* __restrict__ C,
                                                     int M, int N, int K) {
  const int wid  = threadIdx.x >> 5;
  const int lane = threadIdx.x & 31;
  const int mt   = blockIdx.y * 4 + wid;
  const int nt   = blockIdx.x;
  if (mt * 16 >= M) return;
  const int g  = lane >> 4;
  const int lm = lane & 15;
  const int m  = mt * 16 + lm;
  const int n  = nt * 16 + lm;
  const float* Arow = A + m * K;
  v8f acc1 = {}, acc2 = {};
  for (int k0 = 0; k0 < K; k0 += 32) {
    __builtin_prefetch(Arow + k0 + 32, 0, 3);
    v16h af, b1f, b2f;
    load_afrag(af, Arow, k0, g);
#pragma unroll
    for (int h = 0; h < 16; ++h) {
      const int kb = (k0 + b_kidx(h, g)) * N + n;
      b1f[h] = (_Float16)W1[kb];
      b2f[h] = (_Float16)W2[kb];
    }
    acc1 = __builtin_amdgcn_wmma_f32_16x16x32_f16(false, af, false, b1f, (short)0, acc1, false, false);
    acc2 = __builtin_amdgcn_wmma_f32_16x16x32_f16(false, af, false, b2f, (short)0, acc2, false, false);
  }
#pragma unroll
  for (int r = 0; r < 8; ++r) {
    const int mo = mt * 16 + r + 8 * g;
    const int no = nt * 16 + lm;
    const float x1 = acc1[r];
    C[mo * N + no] = x1 * fast_sigmoid(x1) * acc2[r];
  }
}

// ---------------------------------------------------------------------------
// Tiled channel-major transpose for the triangle einsum:
//   dst[c][q] = src[ swap ? ((q&255)*256 + q>>8) : q ][c],  c in [0,256), q in [0,PD)
// ---------------------------------------------------------------------------
__global__ __launch_bounds__(256) void transpose_cm(const float* __restrict__ src,
                                                    float* __restrict__ dst, int swap) {
  __shared__ float tile[32][33];
  const int qb = blockIdx.x * 32;
  const int cb = blockIdx.y * 32;
  const int tx = threadIdx.x;
  for (int yy = threadIdx.y; yy < 32; yy += 8) {
    const int q  = qb + yy;
    const int sp = swap ? ((q & 255) * 256 + (q >> 8)) : q;
    tile[yy][tx] = src[sp * 256 + cb + tx];
  }
  __syncthreads();
  for (int yy = threadIdx.y; yy < 32; yy += 8) {
    dst[(cb + yy) * PD + qb + tx] = tile[tx][yy];
  }
}

// Pad ta_bias_w (128x4) into zero-filled 128x16.
__global__ void pad_bias16(const float* __restrict__ src, float* __restrict__ dst) {
  const int i = blockIdx.x * 256 + threadIdx.x;
  if (i >= 128 * 16) return;
  const int k = i >> 4, n = i & 15;
  dst[i] = (n < 4) ? src[k * 4 + n] : 0.0f;
}
// Pad per-head D=24 -> D=32 with zeros: src 256x(16*24) -> dst 256x(16*32).
__global__ void pad_heads32(const float* __restrict__ src, float* __restrict__ dst) {
  const int i = blockIdx.x * 256 + threadIdx.x;
  if (i >= 256 * 512) return;
  const int row = i >> 9, rem = i & 511, h = rem >> 5, d = rem & 31;
  dst[i] = (d < 24) ? src[row * 384 + h * 24 + d] : 0.0f;
}

// ---------------------------------------------------------------------------
// LayerNorm over last dim; trans=1 reads the z tensor pair-transposed.
// ---------------------------------------------------------------------------
__global__ __launch_bounds__(128) void ln_kernel(const float* __restrict__ x,
                                                 const float* __restrict__ w,
                                                 const float* __restrict__ b,
                                                 float* __restrict__ y,
                                                 int cols, int trans) {
  __shared__ float red[128];
  const int rr = blockIdx.x;
  int ibase;
  if (trans) {
    const int i = rr >> 8, j = rr & 255;
    ibase = (j * 256 + i) * cols;
  } else {
    ibase = rr * cols;
  }
  const int obase = rr * cols;
  const int t = threadIdx.x;
  float s = 0.0f;
  for (int c = t; c < cols; c += 128) s += x[ibase + c];
  red[t] = s; __syncthreads();
  for (int o = 64; o > 0; o >>= 1) { if (t < o) red[t] += red[t + o]; __syncthreads(); }
  const float mean = red[0] / (float)cols;
  __syncthreads();
  float v = 0.0f;
  for (int c = t; c < cols; c += 128) { const float d = x[ibase + c] - mean; v += d * d; }
  red[t] = v; __syncthreads();
  for (int o = 64; o > 0; o >>= 1) { if (t < o) red[t] += red[t + o]; __syncthreads(); }
  const float rstd = rsqrtf(red[0] / (float)cols + 1e-5f);
  for (int c = t; c < cols; c += 128)
    y[obase + c] = (x[ibase + c] - mean) * rstd * w[c] + b[c];
}

// Softmax over rows of length 256 (in place).
__global__ __launch_bounds__(256) void softmax256(float* __restrict__ x) {
  __shared__ float red[256];
  const long long base = (long long)blockIdx.x * 256;
  const int t = threadIdx.x;
  const float v = x[base + t];
  red[t] = v; __syncthreads();
  for (int o = 128; o > 0; o >>= 1) { if (t < o) red[t] = fmaxf(red[t], red[t + o]); __syncthreads(); }
  const float mx = red[0]; __syncthreads();
  const float e = __expf(v - mx);
  red[t] = e; __syncthreads();
  for (int o = 128; o > 0; o >>= 1) { if (t < o) red[t] += red[t + o]; __syncthreads(); }
  x[base + t] = e * __builtin_amdgcn_rcpf(red[0]);
}

__global__ void ew_gate(float* __restrict__ y, const float* __restrict__ p,
                        const float* __restrict__ g, long long n) {
  const long long i = (long long)blockIdx.x * blockDim.x + threadIdx.x;
  if (i < n) y[i] = p[i] * fast_sigmoid(g[i]);
}
__global__ void ew_add_gated(float* __restrict__ z, const float* __restrict__ a,
                             const float* __restrict__ g, long long n) {
  const long long i = (long long)blockIdx.x * blockDim.x + threadIdx.x;
  if (i < n) z[i] += a[i] * fast_sigmoid(g[i]);
}
__global__ void ew_add(float* __restrict__ z, const float* __restrict__ d, long long n) {
  const long long i = (long long)blockIdx.x * blockDim.x + threadIdx.x;
  if (i < n) z[i] += d[i];
}
// z[i,j,c] += d[j,i,c]  (pair-transposed residual add, 256x256xCZ)
__global__ void ew_add_transposed(float* __restrict__ z, const float* __restrict__ d) {
  const long long i = (long long)blockIdx.x * blockDim.x + threadIdx.x;
  if (i >= (long long)PD * CZ) return;
  const int c = (int)(i & (CZ - 1));
  const long long pair = i >> 7;
  const int ii = (int)(pair >> 8), jj = (int)(pair & 255);
  z[i] += d[(((long long)jj * 256 + ii) << 7) + c];
}

// ---------------------------------------------------------------------------
// Host side
// ---------------------------------------------------------------------------
static inline GP gp0() {
  GP p{};
  p.S2 = 1; p.alpha = 1.0f; p.beta = 0.0f;
  return p;
}
static inline void run1(hipStream_t st, const GP& p, int nb, bool bk1) {
  dim3 g((unsigned)((p.N + 15) / 16), (unsigned)((p.M + 63) / 64), (unsigned)nb);
  if (bk1) gemm_fast<true><<<g, dim3(128), 0, st>>>(p);
  else     gemm_fast<false><<<g, dim3(128), 0, st>>>(p);
}
static inline void run2(hipStream_t st, const GP& p, int nb, bool bk1) {
  dim3 g((unsigned)((p.N + 31) / 32), (unsigned)((p.M + 127) / 128), (unsigned)nb);
  if (bk1) gemm_fast2<true><<<g, dim3(128), 0, st>>>(p);
  else     gemm_fast2<false><<<g, dim3(128), 0, st>>>(p);
}

extern "C" void kernel_launch(void* const* d_in, const int* in_sizes, int n_in,
                              void* d_out, int out_size, void* d_ws, size_t ws_size,
                              hipStream_t stream) {
  (void)in_sizes; (void)n_in; (void)out_size; (void)ws_size;
  const float* s_in     = (const float*)d_in[0];
  const float* z_in     = (const float*)d_in[1];
  const float* tm_ni_w  = (const float*)d_in[2];
  const float* tm_ni_b  = (const float*)d_in[3];
  const float* tm_p_in  = (const float*)d_in[4];
  const float* tm_g_in  = (const float*)d_in[5];
  const float* tm_no_w  = (const float*)d_in[6];
  const float* tm_no_b  = (const float*)d_in[7];
  const float* tm_p_out = (const float*)d_in[8];
  const float* tm_g_out = (const float*)d_in[9];
  const float* ta_ln_w  = (const float*)d_in[10];
  const float* ta_ln_b  = (const float*)d_in[11];
  const float* ta_bias_w= (const float*)d_in[12];
  const float* ta_q_w   = (const float*)d_in[13];
  const float* ta_k_w   = (const float*)d_in[14];
  const float* ta_v_w   = (const float*)d_in[15];
  const float* ta_g_w   = (const float*)d_in[16];
  const float* ta_o_w   = (const float*)d_in[17];
  const float* apb_ns_w = (const float*)d_in[18];
  const float* apb_ns_b = (const float*)d_in[19];
  const float* apb_q_w  = (const float*)d_in[20];
  const float* apb_q_b  = (const float*)d_in[21];
  const float* apb_k_w  = (const float*)d_in[22];
  const float* apb_v_w  = (const float*)d_in[23];
  const float* apb_g_w  = (const float*)d_in[24];
  const float* apb_zln_w= (const float*)d_in[25];
  const float* apb_zln_b= (const float*)d_in[26];
  const float* apb_z_w  = (const float*)d_in[27];
  const float* apb_o_w  = (const float*)d_in[28];
  const float* tz_n_w   = (const float*)d_in[29];
  const float* tz_n_b   = (const float*)d_in[30];
  const float* tz_fc1   = (const float*)d_in[31];
  const float* tz_fc2   = (const float*)d_in[32];
  const float* tz_fc3   = (const float*)d_in[33];
  const float* ts_n_w   = (const float*)d_in[34];
  const float* ts_n_b   = (const float*)d_in[35];
  const float* ts_fc1   = (const float*)d_in[36];
  const float* ts_fc2   = (const float*)d_in[37];
  const float* ts_fc3   = (const float*)d_in[38];

  float* s_out = (float*)d_out;              // 256*384
  float* z_out = s_out + (size_t)NRES * CS;  // 256*256*128
  (void)hipMemcpyAsync(s_out, s_in, (size_t)NRES * CS * sizeof(float),
                       hipMemcpyDeviceToDevice, stream);
  (void)hipMemcpyAsync(z_out, z_in, (size_t)PD * CZ * sizeof(float),
                       hipMemcpyDeviceToDevice, stream);

  // workspace layout (floats)
  float* W  = (float*)d_ws;
  float* B1 = W;                         // PD*128 : xn
  float* B2 = B1 + (size_t)PD * CZ;      // PD*256 : pg / q|k   (B2..B3 = PD*512 hidden)
  float* B3 = B2 + (size_t)PD * 256;     // PD*256 : gates / channel-major T / v|g
  float* B4 = B3 + (size_t)PD * 256;     // PD*128 : o / att-out
  float* B5 = B4 + (size_t)PD * CZ;      // PD*128 : on / gate / score chunk / delta
  float* B7 = B5 + (size_t)PD * CZ;      // small region
  float* TB = B7;                        // PD*16  : triangle bias t[i,j,h] (padded)
  float* SN = TB + (size_t)PD * 16;      // 256*384
  float* SQ = SN + (size_t)NRES * CS;
  float* SK = SQ + (size_t)NRES * CS;
  float* SV = SK + (size_t)NRES * CS;
  float* SG = SV + (size_t)NRES * CS;
  float* ZB = SG + (size_t)NRES * CS;    // PD*16  : zb[i,j,h]
  float* AO = ZB + (size_t)PD * 16;      // 256*384
  float* TS = AO + (size_t)NRES * CS;    // 256*1536
  float* WB16 = TS + (size_t)NRES * 1536;// 128*16 : padded ta_bias_w
  float* QP = WB16 + 128 * 16;           // 256*512 : q, head-stride padded to 32
  float* KP = QP + (size_t)NRES * 512;   // 256*512

  auto LN = [&](const float* x, const float* w, const float* b, float* y,
                int rows, int cols, int trans) {
    ln_kernel<<<dim3((unsigned)rows), dim3(128), 0, stream>>>(x, w, b, y, cols, trans);
  };

  // ------------------------- triangle multiply -------------------------
  auto tri_mul = [&](int idx, bool ending) {
    LN(z_out, tm_ni_w + idx * CZ, tm_ni_b + idx * CZ, B1, PD, CZ, 0);
    GP p = gp0();                                   // p = xn @ p_in  (128 -> 256)
    p.A = B1; p.B = tm_p_in + (size_t)idx * CZ * 256; p.C = B2;
    p.M = PD; p.N = 256; p.K = CZ;
    p.sAm = CZ; p.sAk = 1; p.sBk = 256; p.sBn = 1; p.sCm = 256; p.sCn = 1;
    run2(stream, p, 1, false);
    p.B = tm_g_in + (size_t)idx * CZ * 256; p.C = B3;  // gates
    run2(stream, p, 1, false);
    {
      long long n = (long long)PD * 256;
      ew_gate<<<dim3((unsigned)((n + 255) / 256)), 256, 0, stream>>>(B2, B2, B3, n);
    }
    // channel-major transpose: T[c][row][k] (swap folds in the 'ending' case)
    transpose_cm<<<dim3(PD / 32, 256 / 32), dim3(32, 8), 0, stream>>>(B2, B3, ending ? 1 : 0);
    // einsum: 128 independent 256x256x256 GEMMs, both operands K-contiguous
    GP e = gp0();
    e.A = B3; e.B = B3 + (size_t)128 * PD; e.C = B4;
    e.M = 256; e.N = 256; e.K = 256;
    e.sAm = 256; e.sAk = 1; e.sBk = 1; e.sBn = 256;
    e.sCm = 32768; e.sCn = CZ;                       // O[(i*256+j)*128 + c]
    e.bA1 = PD; e.bB1 = PD; e.bC1 = 1;
    run2(stream, e, CZ, true);
    LN(B4, tm_no_w + idx * CZ, tm_no_b + idx * CZ, B5, PD, CZ, 0);
    GP q = gp0();                                   // o2 = LN(o) @ p_out
    q.A = B5; q.B = tm_p_out + (size_t)idx * CZ * CZ; q.C = B4;
    q.M = PD; q.N = CZ; q.K = CZ;
    q.sAm = CZ; q.sAk = 1; q.sBk = CZ; q.sBn = 1; q.sCm = CZ; q.sCn = 1;
    run2(stream, q, 1, false);
    q.A = B1; q.B = tm_g_out + (size_t)idx * CZ * CZ; q.C = B5;  // gate = xn @ g_out
    run2(stream, q, 1, false);
    {
      long long n = (long long)PD * CZ;             // z += o2 * sigmoid(gate)
      ew_add_gated<<<dim3((unsigned)((n + 255) / 256)), 256, 0, stream>>>(z_out, B4, B5, n);
    }
  };

  // ------------------------- triangle attention -------------------------
  auto tri_att = [&](int idx, bool ending) {
    LN(z_out, ta_ln_w + idx * CZ, ta_ln_b + idx * CZ, B1, PD, CZ, ending ? 1 : 0);
    pad_bias16<<<dim3((128 * 16 + 255) / 256), 256, 0, stream>>>(
        ta_bias_w + (size_t)idx * CZ * 4, WB16);
    GP p = gp0();                                   // tb = xn @ padded bias_w (PD x 16)
    p.A = B1; p.B = WB16; p.C = TB;
    p.M = PD; p.N = 16; p.K = CZ;
    p.sAm = CZ; p.sAk = 1; p.sBk = 16; p.sBn = 1; p.sCm = 16; p.sCn = 1;
    run1(stream, p, 1, false);
    float* qb = B2;
    float* kb = B2 + (size_t)PD * CZ;
    float* vb = B3;
    float* gb = B3 + (size_t)PD * CZ;
    const float* wsrc[4] = { ta_q_w + (size_t)idx * CZ * CZ, ta_k_w + (size_t)idx * CZ * CZ,
                             ta_v_w + (size_t)idx * CZ * CZ, ta_g_w + (size_t)idx * CZ * CZ };
    float* dst[4] = { qb, kb, vb, gb };
    for (int t = 0; t < 4; ++t) {
      GP q = gp0();
      q.A = B1; q.B = wsrc[t]; q.C = dst[t];
      q.M = PD; q.N = CZ; q.K = CZ;
      q.sAm = CZ; q.sAk = 1; q.sBk = CZ; q.sBn = 1; q.sCm = CZ; q.sCn = 1;
      run2(stream, q, 1, false);
    }
    const float scale = 0.17677669529663687f;       // 1/sqrt(32)
    for (int rc = 0; rc < 8; ++rc) {                // 32 r-rows x 4 heads per chunk
      const size_t roff = (size_t)rc * 32 * 256 * CZ;
      GP sc = gp0();                                // S = scale * q k^T + tb
      sc.A = qb + roff; sc.B = kb + roff; sc.C = B5; sc.bias = TB;
      sc.M = 256; sc.N = 256; sc.K = 32; sc.S2 = 4;
      sc.sAm = CZ; sc.sAk = 1; sc.sBk = 1; sc.sBn = CZ;
      sc.sCm = 256; sc.sCn = 1;
      sc.bA1 = 256 * CZ; sc.bA2 = 32;
      sc.bB1 = 256 * CZ; sc.bB2 = 32;
      sc.bC1 = 4 * 65536; sc.bC2 = 65536;
      sc.sbm = 256 * 16; sc.sbn = 16; sc.bb1 = 0; sc.bb2 = 1;  // padded TB stride 16
      sc.alpha = scale;
      run2(stream, sc, 32 * 4, true);               // B (k) is K-contiguous
      softmax256<<<dim3(32 * 4 * 256), 256, 0, stream>>>(B5);
      GP pv = gp0();                                // O = P @ V
      pv.A = B5; pv.B = vb + roff; pv.C = B4 + roff;
      pv.M = 256; pv.N = 32; pv.K = 256; pv.S2 = 4;
      pv.sAm = 256; pv.sAk = 1; pv.sBk = CZ; pv.sBn = 1; pv.sCm = CZ; pv.sCn = 1;
      pv.bA1 = 4 * 65536; pv.bA2 = 65536;
      pv.bB1 = 256 * CZ; pv.bB2 = 32;
      pv.bC1 = 256 * CZ; pv.bC2 = 32;
      run2(stream, pv, 32 * 4, false);
    }
    {
      long long n = (long long)PD * CZ;             // o *= sigmoid(g)
      ew_gate<<<dim3((unsigned)((n + 255) / 256)), 256, 0, stream>>>(B4, B4, gb, n);
    }
    GP od = gp0();                                  // delta = o @ o_w
    od.A = B4; od.B = ta_o_w + (size_t)idx * CZ * CZ; od.C = B5;
    od.M = PD; od.N = CZ; od.K = CZ;
    od.sAm = CZ; od.sAk = 1; od.sBk = CZ; od.sBn = 1; od.sCm = CZ; od.sCn = 1;
    run2(stream, od, 1, false);
    const long long n = (long long)PD * CZ;
    if (ending)
      ew_add_transposed<<<dim3((unsigned)((n + 255) / 256)), 256, 0, stream>>>(z_out, B5);
    else
      ew_add<<<dim3((unsigned)((n + 255) / 256)), 256, 0, stream>>>(z_out, B5, n);
  };

  // ------------------------- transitions -------------------------
  auto transition_z = [&]() {
    LN(z_out, tz_n_w, tz_n_b, B1, PD, CZ, 0);
    dim3 g((512 + 15) / 16, (PD + 63) / 64, 1);
    dualgemm_silu<<<g, 128, 0, stream>>>(B1, tz_fc1, tz_fc2, B2, PD, 512, CZ);
    GP p = gp0();                                   // z += t @ fc3
    p.A = B2; p.B = tz_fc3; p.C = z_out;
    p.M = PD; p.N = CZ; p.K = 512;
    p.sAm = 512; p.sAk = 1; p.sBk = CZ; p.sBn = 1; p.sCm = CZ; p.sCn = 1;
    p.beta = 1.0f;
    run2(stream, p, 1, false);
  };
  auto transition_s = [&]() {
    LN(s_out, ts_n_w, ts_n_b, SN, NRES, CS, 0);
    dim3 g((1536 + 15) / 16, (NRES + 63) / 64, 1);
    dualgemm_silu<<<g, 128, 0, stream>>>(SN, ts_fc1, ts_fc2, TS, NRES, 1536, CS);
    GP p = gp0();                                   // s += t @ fc3
    p.A = TS; p.B = ts_fc3; p.C = s_out;
    p.M = NRES; p.N = CS; p.K = 1536;
    p.sAm = 1536; p.sAk = 1; p.sBk = CS; p.sBn = 1; p.sCm = CS; p.sCn = 1;
    p.beta = 1.0f;
    run2(stream, p, 1, false);
  };

  // ------------------------- attention with pair bias -------------------------
  auto apb = [&]() {
    LN(s_out, apb_ns_w, apb_ns_b, SN, NRES, CS, 0);
    GP p = gp0();                                   // q = sn @ q_w + q_b
    p.A = SN; p.B = apb_q_w; p.C = SQ; p.bias = apb_q_b;
    p.M = NRES; p.N = CS; p.K = CS;
    p.sAm = CS; p.sAk = 1; p.sBk = CS; p.sBn = 1; p.sCm = CS; p.sCn = 1;
    p.sbm = 0; p.sbn = 1;
    run2(stream, p, 1, false);
    p.bias = nullptr;
    p.B = apb_k_w; p.C = SK; run2(stream, p, 1, false);
    p.B = apb_v_w; p.C = SV; run2(stream, p, 1, false);
    p.B = apb_g_w; p.C = SG; run2(stream, p, 1, false);
    LN(z_out, apb_zln_w, apb_zln_b, B1, PD, CZ, 0);
    GP zp = gp0();                                  // zb = LN(z) @ z_w  (PD x 16)
    zp.A = B1; zp.B = apb_z_w; zp.C = ZB;
    zp.M = PD; zp.N = 16; zp.K = CZ;
    zp.sAm = CZ; zp.sAk = 1; zp.sBk = 16; zp.sBn = 1; zp.sCm = 16; zp.sCn = 1;
    run1(stream, zp, 1, false);
    // pad q/k to head-stride 32 (zeros in d=24..31) -> K=32 fast path
    pad_heads32<<<dim3((256 * 512 + 255) / 256), 256, 0, stream>>>(SQ, QP);
    pad_heads32<<<dim3((256 * 512 + 255) / 256), 256, 0, stream>>>(SK, KP);
    GP sc = gp0();                                  // S = scale*q k^T + zb (16 heads)
    sc.A = QP; sc.B = KP; sc.C = B5; sc.bias = ZB;
    sc.M = NRES; sc.N = NRES; sc.K = 32; sc.S2 = 1;
    sc.sAm = 512; sc.sAk = 1; sc.sBk = 1; sc.sBn = 512;
    sc.sCm = 256; sc.sCn = 1;
    sc.bA1 = 32; sc.bB1 = 32; sc.bC1 = 65536;
    sc.sbm = 256 * 16; sc.sbn = 16; sc.bb1 = 1;
    sc.alpha = 0.2041241452319315f;                 // 1/sqrt(24)
    run2(stream, sc, 16, true);
    softmax256<<<dim3(16 * 256), 256, 0, stream>>>(B5);
    GP pv = gp0();                                  // O = P @ V (N=24: stores masked)
    pv.A = B5; pv.B = SV; pv.C = AO;
    pv.M = NRES; pv.N = 24; pv.K = NRES; pv.S2 = 1;
    pv.sAm = 256; pv.sAk = 1; pv.sBk = CS; pv.sBn = 1; pv.sCm = CS; pv.sCn = 1;
    pv.bA1 = 65536; pv.bB1 = 24; pv.bC1 = 24;
    run1(stream, pv, 16, false);
    {
      long long n = (long long)NRES * CS;           // o *= sigmoid(sn @ g_w)
      ew_gate<<<dim3((unsigned)((n + 255) / 256)), 256, 0, stream>>>(AO, AO, SG, n);
    }
    GP o = gp0();                                   // s += o @ o_w
    o.A = AO; o.B = apb_o_w; o.C = s_out;
    o.M = NRES; o.N = CS; o.K = CS;
    o.sAm = CS; o.sAk = 1; o.sBk = CS; o.sBn = 1; o.sCm = CS; o.sCn = 1;
    o.beta = 1.0f;
    run2(stream, o, 1, false);
  };

  // ------------------------- full layer -------------------------
  tri_mul(0, false);
  tri_mul(1, true);
  tri_att(0, false);
  tri_att(1, true);
  transition_z();
  apb();
  transition_s();
}
